// RNN_Seq2Seq_39797166965186
// MI455X (gfx1250) — matrix operand order repaired and checked by reading, not compiled, not run on hardware
//
#include <hip/hip_runtime.h>
#include <hip/hip_bf16.h>

typedef __attribute__((ext_vector_type(16))) __bf16         v16bf;
typedef __attribute__((ext_vector_type(8)))  float          v8f;
typedef __attribute__((ext_vector_type(8)))  unsigned short us8;
typedef __attribute__((ext_vector_type(16))) unsigned short us16;

constexpr int EMB   = 150;
constexpr int UNITS = 200;
constexpr int OUT   = 264;
constexpr int BATCH = 100;
constexpr int TLEN  = 1000;
constexpr int GP    = 208;          // per-gate padded width (13 tiles of 16)
constexpr int NP3   = 3 * GP;       // 624
constexpr int KP    = 224;          // padded K for recurrence (200 -> 7 chunks of 32)
constexpr int KE    = 160;          // padded K for embedding GEMM (150 -> 5 chunks)
constexpr int OUTP  = 272;          // padded OUT (17 tiles)
constexpr int KOUT  = 224;          // padded K for output GEMM
constexpr int ROWS  = BATCH * TLEN; // 100000 = 6250 tiles of 16

__device__ inline unsigned short f2bf(float f) {
  unsigned int u = __builtin_bit_cast(unsigned int, f);
  u = (u + 0x7FFFu + ((u >> 16) & 1u)) >> 16;
  return (unsigned short)u;
}
__device__ inline float bf2f(unsigned short h) {
  unsigned int u = ((unsigned int)h) << 16;
  return __builtin_bit_cast(float, u);
}
__device__ inline v8f wmma_bf16(us16 a, us16 b, v8f c) {
  return __builtin_amdgcn_wmma_f32_16x16x32_bf16(
      false, __builtin_bit_cast(v16bf, a),
      false, __builtin_bit_cast(v16bf, b),
      (short)0, c, false, false);
}
__device__ inline float sigm(float x) { return 1.f / (1.f + __expf(-x)); }

// ---------------- Kernel 1: xz = emb[inputs] @ Wx + bx (gate-padded cols) ----
__global__ __launch_bounds__(256) void xz_kernel(const int* __restrict__ inputs,
                                                 const float* __restrict__ emb,
                                                 const float* __restrict__ Wx,
                                                 const float* __restrict__ bx,
                                                 float* __restrict__ xz) {
  __shared__ unsigned short WxT[NP3 * KE];   // [n][k] bf16, ~200 KB
  const int tid = threadIdx.x;
  for (int idx = tid; idx < NP3 * KE; idx += 256) {
    int n = idx / KE, k = idx % KE;
    int g = n / GP, nl = n % GP;
    float v = (nl < UNITS && k < EMB) ? Wx[k * (3 * UNITS) + g * UNITS + nl] : 0.f;
    WxT[idx] = f2bf(v);
  }
  __syncthreads();

  const int wave = tid >> 5, lane = tid & 31;
  const int half = lane >> 4, nl16 = lane & 15;
  const int mt = blockIdx.x * 8 + wave;
  if (mt >= ROWS / 16) return;               // uniform per-wave exit

  const int row = mt * 16 + nl16;            // flat (t*BATCH + b)
  const int t = row / BATCH, b = row % BATCH;
  const long tok = inputs[b * TLEN + t];
  const float* er = emb + tok * EMB;

  us16 A[5];
  #pragma unroll
  for (int c = 0; c < 5; c++) {
    us16 v;
    const int k0 = c * 32 + half * 16;
    #pragma unroll
    for (int hs = 0; hs < 2; hs++) {
      #pragma unroll
      for (int e = 0; e < 8; e++) {
        int k = k0 + hs * 8 + e;
        v[hs * 8 + e] = f2bf((k < EMB) ? er[k] : 0.f);
      }
    }
    A[c] = v;
  }

  for (int nt = 0; nt < 39; nt++) {
    const int ncol = nt * 16 + nl16;
    v8f acc = {};
    #pragma unroll
    for (int c = 0; c < 5; c++) {
      us16 bv = *(const us16*)&WxT[ncol * KE + c * 32 + half * 16];
      acc = wmma_bf16(A[c], bv, acc);
    }
    const int g = ncol / GP, ng = ncol % GP;
    const float bxv = (ng < UNITS) ? bx[g * UNITS + ng] : 0.f;
    #pragma unroll
    for (int i = 0; i < 8; i++) {
      const int r = mt * 16 + i + half * 8;
      xz[(long)r * NP3 + ncol] = acc[i] + bxv;
    }
  }
}

// ---------------- Kernel 2: persistent single-WGP GRU scan -------------------
// 13 waves; wave j owns gate-aligned column tile j (cols j*16..j*16+15 per gate).
// Wh (bf16, gate-padded) lives entirely in registers: 3 gates x 7 k-chunks.
// Hidden state ping-pongs between two bf16 LDS buffers; 1 barrier per step.
__global__ __launch_bounds__(416) void gru_scan(const float* __restrict__ Wh,
                                                const float* __restrict__ bh,
                                                const float* __restrict__ xz,
                                                float* __restrict__ hs) {
  __shared__ unsigned short hbuf[2][112 * KP];   // ~100 KB
  const int tid = threadIdx.x;
  for (int i = tid; i < 2 * 112 * KP; i += 416) ((unsigned short*)hbuf)[i] = 0;

  const int wave = tid >> 5, lane = tid & 31;
  const int half = lane >> 4, nl16 = lane & 15;
  const int j = wave;                 // 0..12
  const int u = j * 16 + nl16;        // owned column
  const bool colv = (u < UNITS);
  const float bhz = colv ? bh[u] : 0.f;
  const float bhr = colv ? bh[UNITS + u] : 0.f;
  const float bhh = colv ? bh[2 * UNITS + u] : 0.f;

  // Stationary B tiles in registers (zeros in all padded regions).
  us16 Bt[3][7];
  #pragma unroll
  for (int g = 0; g < 3; g++) {
    #pragma unroll
    for (int c = 0; c < 7; c++) {
      us16 v;
      const int k0 = c * 32 + half * 16;
      #pragma unroll
      for (int e = 0; e < 16; e++) {
        int k = k0 + e;
        float val = (colv && k < UNITS) ? Wh[k * (3 * UNITS) + g * UNITS + u] : 0.f;
        v[e] = f2bf(val);
      }
      Bt[g][c] = v;
    }
  }
  __syncthreads();

  for (int t = 0; t < TLEN; t++) {
    const unsigned short* hr = hbuf[t & 1];
    unsigned short* hw = hbuf[(t + 1) & 1];
    const float* xzt = xz + (long)t * BATCH * NP3;

    for (int m = 0; m < 7; m++) {
      v8f az = {}, ar = {}, ah = {};
      const int arow = m * 16 + nl16;
      #pragma unroll
      for (int c = 0; c < 7; c++) {
        const int k0 = c * 32 + half * 16;
        us8 lo = *(const us8*)&hr[arow * KP + k0];
        us8 hi = *(const us8*)&hr[arow * KP + k0 + 8];
        us16 av;
        #pragma unroll
        for (int e = 0; e < 8; e++) { av[e] = lo[e]; av[8 + e] = hi[e]; }
        az = wmma_bf16(av, Bt[0][c], az);
        ar = wmma_bf16(av, Bt[1][c], ar);
        ah = wmma_bf16(av, Bt[2][c], ah);
      }
      #pragma unroll
      for (int i = 0; i < 8; i++) {
        const int r = m * 16 + i + half * 8;    // batch row of this acc element
        const bool valid = (r < BATCH) && colv;
        float xzz = 0.f, xzr = 0.f, xzh = 0.f, hold = 0.f;
        if (valid) {
          const float* xr = xzt + (long)r * NP3;
          xzz = xr[u];
          xzr = xr[GP + u];
          xzh = xr[2 * GP + u];
          hold = bf2f(hr[r * KP + u]);
        }
        float z  = sigm(xzz + az[i] + bhz);
        float rr = sigm(xzr + ar[i] + bhr);
        float hh = tanhf(xzh + rr * (ah[i] + bhh));
        float hn = z * hold + (1.f - z) * hh;
        if (!valid) hn = 0.f;
        hw[r * KP + u] = f2bf(hn);
        if (valid) hs[((long)r * TLEN + t) * UNITS + u] = hn;
      }
    }
    __syncthreads();
  }
}

// ---------------- Kernel 3: logits = hs @ Wd + bd, row softmax ---------------
__global__ __launch_bounds__(256) void out_kernel(const float* __restrict__ hs,
                                                  const float* __restrict__ Wd,
                                                  const float* __restrict__ bd,
                                                  float* __restrict__ out) {
  __shared__ unsigned short WdT[OUTP * KOUT];  // ~122 KB
  __shared__ float lg[16 * OUTP];              // ~17 KB
  const int tid = threadIdx.x;
  for (int idx = tid; idx < OUTP * KOUT; idx += 256) {
    int n = idx / KOUT, k = idx % KOUT;
    WdT[idx] = f2bf((n < OUT && k < UNITS) ? Wd[k * OUT + n] : 0.f);
  }
  __syncthreads();

  const int wave = tid >> 5, lane = tid & 31;
  const int half = lane >> 4, nl16 = lane & 15;
  const int mt = blockIdx.x;
  const int row = mt * 16 + nl16;              // flat (b*TLEN + t)
  const float* hrow = hs + (long)row * UNITS;

  us16 A[7];
  #pragma unroll
  for (int c = 0; c < 7; c++) {
    us16 v;
    const int k0 = c * 32 + half * 16;
    #pragma unroll
    for (int hsel = 0; hsel < 2; hsel++) {
      #pragma unroll
      for (int e = 0; e < 8; e++) {
        int k = k0 + hsel * 8 + e;
        v[hsel * 8 + e] = f2bf((k < UNITS) ? hrow[k] : 0.f);
      }
    }
    A[c] = v;
  }

  for (int nt = wave; nt < 17; nt += 8) {
    const int ncol = nt * 16 + nl16;
    v8f acc = {};
    #pragma unroll
    for (int c = 0; c < 7; c++) {
      us16 bv = *(const us16*)&WdT[ncol * KOUT + c * 32 + half * 16];
      acc = wmma_bf16(A[c], bv, acc);
    }
    const float bdv = (ncol < OUT) ? bd[ncol] : 0.f;
    #pragma unroll
    for (int i = 0; i < 8; i++) lg[(i + half * 8) * OUTP + ncol] = acc[i] + bdv;
  }
  __syncthreads();

  // Softmax: 16 threads per row (lanes 0-15 / 16-31 of each wave handle 2 rows)
  const int r = tid >> 4, c0 = tid & 15;
  float mx = -3.4e38f;
  for (int c = c0; c < OUT; c += 16) mx = fmaxf(mx, lg[r * OUTP + c]);
  #pragma unroll
  for (int s = 8; s >= 1; s >>= 1) mx = fmaxf(mx, __shfl_xor(mx, s, 32));
  float sum = 0.f;
  for (int c = c0; c < OUT; c += 16) {
    float e = __expf(lg[r * OUTP + c] - mx);
    lg[r * OUTP + c] = e;
    sum += e;
  }
  #pragma unroll
  for (int s = 8; s >= 1; s >>= 1) sum += __shfl_xor(sum, s, 32);
  const float inv = 1.f / sum;
  const long orow = (long)(mt * 16 + r) * OUT;
  for (int c = c0; c < OUT; c += 16) out[orow + c] = lg[r * OUTP + c] * inv;
}

// ---------------- Launch -----------------------------------------------------
extern "C" void kernel_launch(void* const* d_in, const int* in_sizes, int n_in,
                              void* d_out, int out_size, void* d_ws, size_t ws_size,
                              hipStream_t stream) {
  const int*   inputs = (const int*)d_in[0];
  const float* emb    = (const float*)d_in[1];
  const float* Wx     = (const float*)d_in[2];
  const float* Wh     = (const float*)d_in[3];
  const float* bx     = (const float*)d_in[4];
  const float* bh     = (const float*)d_in[5];
  const float* Wd     = (const float*)d_in[6];
  const float* bd     = (const float*)d_in[7];
  float* out = (float*)d_out;

  char* ws = (char*)d_ws;
  float* xz = (float*)ws;                               // ROWS*NP3 f32 = 249.6 MB
  float* hsbuf = (float*)(ws + (size_t)ROWS * NP3 * 4); // ROWS*UNITS f32 = 80 MB

  xz_kernel<<<(ROWS / 16 + 7) / 8, 256, 0, stream>>>(inputs, emb, Wx, bx, xz);
  gru_scan<<<1, 416, 0, stream>>>(Wh, bh, xz, hsbuf);
  out_kernel<<<ROWS / 16, 256, 0, stream>>>(hsbuf, Wd, bd, out);
}
